// CrossAttention_28845000360186
// MI455X (gfx1250) — compile-verified, matrix operand order checked by reading, not run on hardware
//
#include <hip/hip_runtime.h>
#include <hip/hip_bf16.h>

typedef __attribute__((ext_vector_type(16))) __bf16        v16bf;
typedef __attribute__((ext_vector_type(8)))  float         v8f;
typedef __attribute__((ext_vector_type(8)))  unsigned int  v8u;
typedef __attribute__((ext_vector_type(4)))  unsigned int  v4u;
typedef __attribute__((ext_vector_type(8)))  int           v8i;
typedef __attribute__((ext_vector_type(4)))  int           v4i;

#define SEQ   2048
#define DIM   1024
#define HEADS 16
#define HD    64
#define MTOT  4096   // b*n
#define SCALE 0.125f // 64^-0.5

// ---------- helpers ----------
__device__ inline unsigned short f2bf(float f) {
    unsigned int u = __float_as_uint(f);
    unsigned int r = u + 0x7FFFu + ((u >> 16) & 1u);  // round-to-nearest-even
    return (unsigned short)(r >> 16);
}
__device__ inline unsigned int pack2bf(float lo, float hi) {
    return (unsigned int)f2bf(lo) | ((unsigned int)f2bf(hi) << 16);
}
__device__ inline v8f wmma_bf16(v16bf a, v16bf b, v8f c) {
    return __builtin_amdgcn_wmma_f32_16x16x32_bf16(false, a, false, b, (short)0, c, false, false);
}
// A/B fragment load (16-bit A-matrix 16x32 layout): lanes 0-15 -> K {0-7,16-23},
// lanes 16-31 -> K {8-15,24-31}; two contiguous 16B reads per lane.
__device__ inline v16bf load_frag(const unsigned short* p, int row0, int ld, int col0) {
    int lane = threadIdx.x & 31;
    int r  = row0 + (lane & 15);
    int kb = col0 + ((lane >> 4) << 3);
    const uint4* q0 = (const uint4*)(p + (size_t)r * ld + kb);
    const uint4* q1 = (const uint4*)(p + (size_t)r * ld + kb + 16);
    uint4 a = *q0, b = *q1;
    v8u u;
    u[0] = a.x; u[1] = a.y; u[2] = a.z; u[3] = a.w;
    u[4] = b.x; u[5] = b.y; u[6] = b.z; u[7] = b.w;
    return __builtin_bit_cast(v16bf, u);
}
__device__ inline float lane_read(float v, int srcLane) {
    return __uint_as_float((unsigned)__builtin_amdgcn_ds_bpermute(srcLane << 2, (int)__float_as_uint(v)));
}

// ---------- TDM: 2-D bf16 tile (tile_d1 rows x tile_d0 elems) -> LDS ----------
// D# packing per CDNA5 ISA. data_size = 2 bytes.
__device__ inline void tdm_load_2d(unsigned lds_off, const unsigned short* gptr,
                                   unsigned tensor_d0, unsigned tensor_d1,
                                   unsigned long long stride0,
                                   unsigned tile_d0, unsigned tile_d1) {
    unsigned long long ga = (unsigned long long)(size_t)gptr;
    v4u g0;
    g0[0] = 1u;                                                // count=1, user mode
    g0[1] = lds_off;                                           // LDS byte address
    g0[2] = (unsigned)(ga & 0xFFFFFFFFu);                      // global_addr lo
    g0[3] = (unsigned)((ga >> 32) & 0x1FFFFFFu) | (2u << 30);  // addr hi | type=2
    v8i g1;
    g1[0] = (int)(1u << 16);                                   // data_size=2B
    g1[1] = (int)((tensor_d0 & 0xFFFFu) << 16);                // [79:48] tensor_dim0
    g1[2] = (int)((tensor_d0 >> 16) | ((tensor_d1 & 0xFFFFu) << 16));
    g1[3] = (int)((tensor_d1 >> 16) | (tile_d0 << 16));        // tile_dim0
    g1[4] = (int)tile_d1;                                      // tile_dim1 (tile_dim2=0)
    g1[5] = (int)(stride0 & 0xFFFFFFFFu);                      // tensor_dim0_stride
    g1[6] = (int)((stride0 >> 32) & 0xFFFFu);                  // stride1 = 0
    g1[7] = 0;
    v4i g2 = {0, 0, 0, 0};
    v4i g3 = {0, 0, 0, 0};
#if defined(__clang_major__) && __clang_major__ >= 23
    v8i g4 = {0, 0, 0, 0, 0, 0, 0, 0};
    __builtin_amdgcn_tensor_load_to_lds(g0, g1, g2, g3, g4, 0);
#else
    __builtin_amdgcn_tensor_load_to_lds(g0, g1, g2, g3, 0);
#endif
}

// ---------- fp32 -> bf16 row-major conversion (x, context) ----------
__global__ __launch_bounds__(256) void to_bf16(const float* __restrict__ in,
                                               unsigned short* __restrict__ out) {
    int i = blockIdx.x * 256 + threadIdx.x;      // 8 elems per thread
    const float4* p = (const float4*)in;
    float4 a = p[2 * i], b = p[2 * i + 1];
    ((uint4*)out)[i] = make_uint4(pack2bf(a.x, a.y), pack2bf(a.z, a.w),
                                  pack2bf(b.x, b.y), pack2bf(b.z, b.w));
}

// ---------- fp32 [k][n] -> bf16 transposed [n][k] (weights) ----------
__global__ __launch_bounds__(256) void transpose_w(const float* __restrict__ W,
                                                   unsigned short* __restrict__ Wt) {
    __shared__ float tile[32][33];
    int k0 = blockIdx.y * 32, n0 = blockIdx.x * 32;
    int t = threadIdx.x;
    int r = t >> 3, c = (t & 7) * 4;
    float4 v = *(const float4*)(W + (size_t)(k0 + r) * DIM + n0 + c);
    tile[r][c] = v.x; tile[r][c + 1] = v.y; tile[r][c + 2] = v.z; tile[r][c + 3] = v.w;
    __syncthreads();
    uint2 o;
    o.x = pack2bf(tile[c][r], tile[c + 1][r]);
    o.y = pack2bf(tile[c + 2][r], tile[c + 3][r]);
    *(uint2*)(Wt + (size_t)(n0 + r) * DIM + k0 + c) = o;
}

// ---------- GEMM: C = A(bf16 [MTOT][DIM]) @ Wt(bf16 [n][k]) + bias ----------
// TDM double-buffered LDS staging; wave 0 drives the DMA pipeline.
// mode 0: bf16 out -> [b,h,n,hd]   (Q,K)
// mode 1: bf16 out -> [b,h,hd,n]   (V transposed for attention B-fragments)
// mode 2: fp32 out -> row-major, += residual (output projection)
__global__ __launch_bounds__(256) void gemm_bf(
    const unsigned short* __restrict__ A, const unsigned short* __restrict__ Wt,
    const float* __restrict__ bias, int mode,
    unsigned short* __restrict__ outB, float* __restrict__ outF,
    const float* __restrict__ resid)
{
    __shared__ unsigned short As[2][128 * 32];   // 2 x 8KB
    __shared__ unsigned short Ws[2][128 * 32];
    int tid  = threadIdx.x;
    int lane = tid & 31;
    int wid  = tid >> 5;
    int row0 = blockIdx.y * 128;
    int col0 = blockIdx.x * 128;
    int wm = wid >> 1, wn = wid & 1;             // 4x2 wave grid, wave tile 32x64
    v8f acc[2][4] = {};

    const unsigned short* Abase = A  + (size_t)row0 * DIM;
    const unsigned short* Wbase = Wt + (size_t)col0 * DIM;
    unsigned ldsA[2], ldsW[2];
    ldsA[0] = (unsigned)(size_t)(void*)As[0];
    ldsA[1] = (unsigned)(size_t)(void*)As[1];
    ldsW[0] = (unsigned)(size_t)(void*)Ws[0];
    ldsW[1] = (unsigned)(size_t)(void*)Ws[1];

    const int NT = DIM / 32;
    if (wid == 0) {                              // preload tile 0
        tdm_load_2d(ldsA[0], Abase, DIM, MTOT, DIM, 32, 128);
        tdm_load_2d(ldsW[0], Wbase, DIM, DIM,  DIM, 32, 128);
        __builtin_amdgcn_s_wait_tensorcnt(0);
    }
    __syncthreads();

    for (int it = 0; it < NT; ++it) {
        int cur = it & 1;
        if (wid == 0 && it + 1 < NT) {           // prefetch next tile into other buffer
            int kb = (it + 1) * 32;
            tdm_load_2d(ldsA[cur ^ 1], Abase + kb, DIM, MTOT, DIM, 32, 128);
            tdm_load_2d(ldsW[cur ^ 1], Wbase + kb, DIM, DIM,  DIM, 32, 128);
        }
        v16bf af[2], bfr[4];
#pragma unroll
        for (int im = 0; im < 2; ++im) af[im]  = load_frag(As[cur], wm * 32 + im * 16, 32, 0);
#pragma unroll
        for (int jn = 0; jn < 4; ++jn) bfr[jn] = load_frag(Ws[cur], wn * 64 + jn * 16, 32, 0);
#pragma unroll
        for (int im = 0; im < 2; ++im)
#pragma unroll
            for (int jn = 0; jn < 4; ++jn)
                acc[im][jn] = wmma_bf16(af[im], bfr[jn], acc[im][jn]);
        if (wid == 0 && it + 1 < NT)
            __builtin_amdgcn_s_wait_tensorcnt(0);
        __syncthreads();
    }

    // epilogue: C/D layout -> lane = col (lane&15), vgpr r = row (r + 8*half)
    int hl = (lane >> 4) & 1;
#pragma unroll
    for (int im = 0; im < 2; ++im) {
#pragma unroll
        for (int jn = 0; jn < 4; ++jn) {
            int gc = col0 + wn * 64 + jn * 16 + (lane & 15);
            float bi = bias[gc];
            int gm0 = row0 + wm * 32 + im * 16 + hl * 8;
#pragma unroll
            for (int r = 0; r < 8; ++r) {
                int gm = gm0 + r;
                float val = acc[im][jn][r] + bi;
                if (mode == 2) {
                    size_t idx = (size_t)gm * DIM + gc;
                    outF[idx] = val + resid[idx];
                } else {
                    int bb = gm >> 11, nrow = gm & (SEQ - 1);
                    int h = gc >> 6, hd = gc & (HD - 1);
                    size_t idx;
                    if (mode == 0)
                        idx = (((size_t)(bb * HEADS + h)) * SEQ + nrow) * HD + hd;
                    else
                        idx = (((size_t)(bb * HEADS + h)) * HD + hd) * SEQ + nrow;
                    outB[idx] = f2bf(val);
                }
            }
        }
    }
}

// ---------- Flash attention: one wave = 16 query rows, no LDS ----------
// S^T = K @ Q^T so softmax is in-lane; S^T accumulator layout == A-operand
// layout of P for the P@V WMMA (no transpose round-trip).
__global__ __launch_bounds__(128) void attn_kernel(
    const unsigned short* __restrict__ Q,   // [b,h,n,hd] bf16
    const unsigned short* __restrict__ Kb,  // [b,h,n,hd] bf16
    const unsigned short* __restrict__ Vt,  // [b,h,hd,n] bf16
    unsigned short* __restrict__ O)         // [b*n, DIM] bf16
{
    int tid  = threadIdx.x;
    int lane = tid & 31;
    int w    = blockIdx.x * 4 + (tid >> 5);
    int bh   = w >> 7;                      // 128 q-tiles per (b,h)
    int qt   = w & 127;
    const unsigned short* Qp = Q  + (size_t)bh * SEQ * HD;
    const unsigned short* Kp = Kb + (size_t)bh * SEQ * HD;
    const unsigned short* Vp = Vt + (size_t)bh * HD * SEQ;
    int qrow = qt * 16;

    v16bf qf0 = load_frag(Qp, qrow, HD, 0);     // used as B operand (Q^T)
    v16bf qf1 = load_frag(Qp, qrow, HD, 32);

    v8f o[4] = {};
    float m_run = -1e30f, l_run = 0.f;
    int hb8   = (lane & 16) >> 1;               // 8 on upper half
    int xlane = lane ^ 16;

    for (int kc = 0; kc < SEQ; kc += 64) {
        v8f s[4];
#pragma unroll
        for (int t = 0; t < 4; ++t) {
            v16bf k0 = load_frag(Kp, kc + t * 16, HD, 0);
            v16bf k1 = load_frag(Kp, kc + t * 16, HD, 32);
            v8f z = {};
            s[t] = wmma_bf16(k0, qf0, z);
            s[t] = wmma_bf16(k1, qf1, s[t]);
        }
        float mc = -1e30f;
#pragma unroll
        for (int t = 0; t < 4; ++t)
#pragma unroll
            for (int r = 0; r < 8; ++r) {
                s[t][r] *= SCALE;
                mc = fmaxf(mc, s[t][r]);
            }
        mc = fmaxf(mc, lane_read(mc, xlane));
        float mn   = fmaxf(m_run, mc);
        float corr = __expf(m_run - mn);
        m_run = mn;
        float ls = 0.f;
#pragma unroll
        for (int t = 0; t < 4; ++t)
#pragma unroll
            for (int r = 0; r < 8; ++r) {
                s[t][r] = __expf(s[t][r] - mn);
                ls += s[t][r];
            }
        ls += lane_read(ls, xlane);
        l_run = l_run * corr + ls;
        float cr[8];
#pragma unroll
        for (int r = 0; r < 8; ++r) cr[r] = lane_read(corr, r + hb8);
#pragma unroll
        for (int t = 0; t < 4; ++t)
#pragma unroll
            for (int r = 0; r < 8; ++r) o[t][r] *= cr[r];
        v8u u0, u1;
#pragma unroll
        for (int j = 0; j < 4; ++j) {
            u0[j]     = pack2bf(s[0][2 * j], s[0][2 * j + 1]);
            u0[j + 4] = pack2bf(s[1][2 * j], s[1][2 * j + 1]);
            u1[j]     = pack2bf(s[2][2 * j], s[2][2 * j + 1]);
            u1[j + 4] = pack2bf(s[3][2 * j], s[3][2 * j + 1]);
        }
        v16bf pa0 = __builtin_bit_cast(v16bf, u0);
        v16bf pa1 = __builtin_bit_cast(v16bf, u1);
#pragma unroll
        for (int t = 0; t < 4; ++t) {
            v16bf v0 = load_frag(Vp, t * 16, SEQ, kc);
            v16bf v1 = load_frag(Vp, t * 16, SEQ, kc + 32);
            o[t] = wmma_bf16(pa0, v0, o[t]);
            o[t] = wmma_bf16(pa1, v1, o[t]);
        }
    }
    float li = 1.f / l_run;
    float lr[8];
#pragma unroll
    for (int r = 0; r < 8; ++r) lr[r] = lane_read(li, r + hb8);
    int b = bh >> 4, h = bh & (HEADS - 1);
    int rowb = b * SEQ + qrow + ((lane >> 4) & 1) * 8;
    int colb = h * HD + (lane & 15);
#pragma unroll
    for (int t = 0; t < 4; ++t)
#pragma unroll
        for (int r = 0; r < 8; ++r)
            O[(size_t)(rowb + r) * DIM + colb + t * 16] = f2bf(o[t][r] * lr[r]);
}

// ---------- LayerNorm over d=1024, one block per row ----------
__global__ __launch_bounds__(256) void layernorm_k(
    const float* __restrict__ y, const float* __restrict__ g,
    const float* __restrict__ be, float* __restrict__ out)
{
    __shared__ float red[256];
    int row = blockIdx.x, t = threadIdx.x;
    const float* yr = y + (size_t)row * DIM;
    float v0 = yr[t], v1 = yr[t + 256], v2 = yr[t + 512], v3 = yr[t + 768];
    red[t] = v0 + v1 + v2 + v3;
    __syncthreads();
    for (int off = 128; off > 0; off >>= 1) {
        if (t < off) red[t] += red[t + off];
        __syncthreads();
    }
    float mu = red[0] * (1.f / DIM);
    __syncthreads();
    float d0 = v0 - mu, d1 = v1 - mu, d2 = v2 - mu, d3 = v3 - mu;
    red[t] = d0 * d0 + d1 * d1 + d2 * d2 + d3 * d3;
    __syncthreads();
    for (int off = 128; off > 0; off >>= 1) {
        if (t < off) red[t] += red[t + off];
        __syncthreads();
    }
    float inv = rsqrtf(red[0] * (1.f / DIM) + 1e-5f);
    float* orow = out + (size_t)row * DIM;
    orow[t]       = d0 * inv * g[t]       + be[t];
    orow[t + 256] = d1 * inv * g[t + 256] + be[t + 256];
    orow[t + 512] = d2 * inv * g[t + 512] + be[t + 512];
    orow[t + 768] = d3 * inv * g[t + 768] + be[t + 768];
}

extern "C" void kernel_launch(void* const* d_in, const int* in_sizes, int n_in,
                              void* d_out, int out_size, void* d_ws, size_t ws_size,
                              hipStream_t stream) {
    const float* x     = (const float*)d_in[0];
    const float* ctx   = (const float*)d_in[1];
    const float* Wq    = (const float*)d_in[2];
    const float* bq    = (const float*)d_in[3];
    const float* Wk    = (const float*)d_in[4];
    const float* bk    = (const float*)d_in[5];
    const float* Wv    = (const float*)d_in[6];
    const float* bv    = (const float*)d_in[7];
    const float* Wo    = (const float*)d_in[8];
    const float* bo    = (const float*)d_in[9];
    const float* gamma = (const float*)d_in[10];
    const float* beta  = (const float*)d_in[11];

    char* ws = (char*)d_ws;
    unsigned short* qb  = (unsigned short*)(ws);                      // 8 MB [b,h,n,hd]
    unsigned short* kb  = (unsigned short*)(ws + (size_t)( 8 << 20)); // 8 MB
    unsigned short* vt  = (unsigned short*)(ws + (size_t)(16 << 20)); // 8 MB [b,h,hd,n]
    unsigned short* aob = (unsigned short*)(ws + (size_t)(24 << 20)); // 8 MB bf16 attn out
    unsigned short* xbf = (unsigned short*)(ws + (size_t)(32 << 20)); // 8 MB
    unsigned short* cbf = (unsigned short*)(ws + (size_t)(40 << 20)); // 8 MB
    unsigned short* wqt = (unsigned short*)(ws + (size_t)(48 << 20)); // 2 MB each
    unsigned short* wkt = (unsigned short*)(ws + (size_t)(50 << 20));
    unsigned short* wvt = (unsigned short*)(ws + (size_t)(52 << 20));
    unsigned short* wot = (unsigned short*)(ws + (size_t)(54 << 20));
    float*          yy  = (float*)         (ws + (size_t)(56 << 20)); // 16 MB

    // stage 0: precision prep (bf16 activations, bf16 transposed weights)
    to_bf16<<<(MTOT * DIM) / (256 * 8), 256, 0, stream>>>(x,   xbf);
    to_bf16<<<(MTOT * DIM) / (256 * 8), 256, 0, stream>>>(ctx, cbf);
    dim3 tg(DIM / 32, DIM / 32);
    transpose_w<<<tg, 256, 0, stream>>>(Wq, wqt);
    transpose_w<<<tg, 256, 0, stream>>>(Wk, wkt);
    transpose_w<<<tg, 256, 0, stream>>>(Wv, wvt);
    transpose_w<<<tg, 256, 0, stream>>>(Wo, wot);

    // stage 1: projections (TDM-fed WMMA GEMMs)
    dim3 gg(DIM / 128, MTOT / 128), bb(256);
    gemm_bf<<<gg, bb, 0, stream>>>(xbf, wqt, bq, 0, qb, nullptr, nullptr);
    gemm_bf<<<gg, bb, 0, stream>>>(cbf, wkt, bk, 0, kb, nullptr, nullptr);
    gemm_bf<<<gg, bb, 0, stream>>>(cbf, wvt, bv, 1, vt, nullptr, nullptr);

    // stage 2: attention
    attn_kernel<<<(2 * HEADS * (SEQ / 16)) / 4, 128, 0, stream>>>(qb, kb, vt, aob);

    // stage 3: output projection + residual, then layernorm
    gemm_bf<<<gg, bb, 0, stream>>>(aob, wot, bo, 2, nullptr, yy, x);
    layernorm_k<<<MTOT, 256, 0, stream>>>(yy, gamma, beta, (float*)d_out);
}
